// Attention_55576876810478
// MI455X (gfx1250) — compile-verified, matrix operand order checked by reading, not run on hardware
//
#include <hip/hip_runtime.h>
#include <hip/hip_bf16.h>
#include <math.h>

// ---- problem dims (match reference) ----
constexpr int Bb = 4, Qq = 1024, Kk = 1024, Dd = 256, Hh = 8, Cc = 32;
constexpr int HC = Hh * Cc; // 256

typedef __attribute__((ext_vector_type(16))) __bf16 v16bf;
typedef __attribute__((ext_vector_type(8)))  float  v8f;

// A-fragment (and B-fragment) k-index for 16x16x32 bf16 WMMA:
// lane 0-15: K = 0..7 (i=0..7), 16..23 (i=8..15)
// lane16-31: K = 8..15,         24..31
__device__ __forceinline__ int akidx(int half, int i) {
  return (i < 8) ? (half * 8 + i) : (8 + half * 8 + i);
}

__device__ __forceinline__ v8f wmma_bf16(const v16bf& a, const v16bf& b, const v8f& c) {
  return __builtin_amdgcn_wmma_f32_16x16x32_bf16(false, a, false, b, (short)0, c, false, false);
}

// -----------------------------------------------------------------------------
// Generic 16x16-output-tile WMMA GEMM over Kd (multiple of 32).
// MODE 0: out_bf16 = acc * scale            (q projection, scale = 1/sqrt(C))
// MODE 1: out_bf16 = acc                    (k / v projections)
// MODE 3: out_f32  = sigmoid(acc + bias[n]) (gate)
// MODE 4: A is bf16; out_f32 = acc + bias[n] (output projection)
// block = 128 threads (4 waves, 4 adjacent column tiles), grid = (M/16, N/64)
// -----------------------------------------------------------------------------
template <int MODE>
__global__ void gemm16_wmma(const void* __restrict__ Xp,
                            const float* __restrict__ W,
                            const float* __restrict__ bias,
                            void* __restrict__ outp,
                            int M, int N, int Kd, float scale) {
  const int lane = threadIdx.x & 31;
  const int wave = threadIdx.x >> 5;
  const int half = lane >> 4;
  const int ln   = lane & 15;
  const int m0 = blockIdx.x * 16;
  const int n0 = (blockIdx.y * 4 + wave) * 16;
  if (m0 >= M || n0 >= N) return;

  v8f acc = {};
  for (int k0 = 0; k0 < Kd; k0 += 32) {
    v16bf a, b;
#pragma unroll
    for (int i = 0; i < 16; ++i) {
      const int kc = k0 + akidx(half, i);
      if (MODE == 4) a[i] = ((const __bf16*)Xp)[(size_t)(m0 + ln) * Kd + kc];
      else           a[i] = (__bf16)(((const float*)Xp)[(size_t)(m0 + ln) * Kd + kc]);
      b[i] = (__bf16)(W[(size_t)kc * N + (n0 + ln)]);
    }
    acc = wmma_bf16(a, b, acc);
  }

#pragma unroll
  for (int j = 0; j < 8; ++j) {
    const int row = m0 + j + 8 * half;
    const int col = n0 + ln;
    float v = acc[j];
    if (MODE == 0) {
      ((__bf16*)outp)[(size_t)row * N + col] = (__bf16)(v * scale);
    } else if (MODE == 1) {
      ((__bf16*)outp)[(size_t)row * N + col] = (__bf16)v;
    } else if (MODE == 3) {
      v = 1.0f / (1.0f + __expf(-(v + bias[col])));
      ((float*)outp)[(size_t)row * N + col] = v;
    } else { // MODE 4
      ((float*)outp)[(size_t)row * N + col] = v + bias[col];
    }
  }
}

// -----------------------------------------------------------------------------
// Flash attention with pair bias + mask bias + gating epilogue.
// One wave per (b, h, 16-row q tile). kv stepped by 32 (two 16-col WMMAs for
// scores, two WMMAs for P*V with K=32 bf16). Online softmax in registers;
// P redistributed D-layout -> A-layout through 1KB LDS.
// The 128MB bias_pair stream bounds the whole pipeline (23.3 TB/s HBM), so we
// explicitly prefetch the next kv-step's 16x32 f32 bias tile (one 64B segment
// per lane -> global_prefetch_b8) while the current step computes.
// grid = (Q/16, B*H), block = 32 (one wave32)
// -----------------------------------------------------------------------------
__global__ void attn_flash_wmma(const __bf16* __restrict__ qw,
                                const __bf16* __restrict__ kw,
                                const __bf16* __restrict__ vw,
                                const float*  __restrict__ gate,
                                const float*  __restrict__ bias_mask,
                                const float*  __restrict__ bias_pair,
                                __bf16* __restrict__ ow) {
  __shared__ __bf16 pl[16 * 32];

  const int lane = threadIdx.x & 31;
  const int half = lane >> 4;
  const int ln   = lane & 15;
  const int q0 = blockIdx.x * 16;
  const int bh = blockIdx.y;
  const int b = bh / Hh, h = bh % Hh;

  const float* bias_base = bias_pair + (size_t)(b * Hh + h) * Qq * Kk;

  // Q A-fragment: row m = q0 + ln, contraction dim c = akidx
  v16bf qa;
#pragma unroll
  for (int i = 0; i < 16; ++i) {
    const int c = akidx(half, i);
    qa[i] = qw[(size_t)(b * Qq + q0 + ln) * HC + h * Cc + c];
  }

  float mi[8], li[8];
  v8f o0 = {}, o1 = {};
#pragma unroll
  for (int j = 0; j < 8; ++j) { mi[j] = -INFINITY; li[j] = 0.0f; }

  for (int kv0 = 0; kv0 < Kk; kv0 += 32) {
    // prefetch next kv step's bias_pair tile: 16 rows x 128B, one 64B seg/lane
    if (kv0 + 32 < Kk) {
      const int prow = q0 + (lane >> 1);
      const int pcol = kv0 + 32 + (lane & 1) * 16;
      __builtin_prefetch(bias_base + (size_t)prow * Kk + pcol, 0, 1);
    }

    // K B-fragments: n = kv column, k = c
    v16bf kb0, kb1;
#pragma unroll
    for (int i = 0; i < 16; ++i) {
      const int c = akidx(half, i);
      kb0[i] = kw[(size_t)(b * Kk + kv0 + ln) * HC + h * Cc + c];
      kb1[i] = kw[(size_t)(b * Kk + kv0 + 16 + ln) * HC + h * Cc + c];
    }
    v8f s0 = {}, s1 = {};
    s0 = wmma_bf16(qa, kb0, s0);
    s1 = wmma_bf16(qa, kb1, s1);

    // additive biases in C/D layout (row = q0+j+8*half, col = ln / 16+ln)
    const float bm0 = bias_mask[(size_t)b * Kk + kv0 + ln];
    const float bm1 = bias_mask[(size_t)b * Kk + kv0 + 16 + ln];
    float p0[8], p1[8];
#pragma unroll
    for (int j = 0; j < 8; ++j) {
      const int row = q0 + j + 8 * half;
      const float* bp = bias_base + (size_t)row * Kk + kv0;
      s0[j] += bm0 + bp[ln];
      s1[j] += bm1 + bp[16 + ln];
    }

    // online softmax: row reductions across the 16-lane half-group
#pragma unroll
    for (int j = 0; j < 8; ++j) {
      float t = fmaxf(s0[j], s1[j]);
      t = fmaxf(t, __shfl_xor(t, 1, 32));
      t = fmaxf(t, __shfl_xor(t, 2, 32));
      t = fmaxf(t, __shfl_xor(t, 4, 32));
      t = fmaxf(t, __shfl_xor(t, 8, 32));
      const float mnew = fmaxf(mi[j], t);
      const float sc = __expf(mi[j] - mnew);
      p0[j] = __expf(s0[j] - mnew);
      p1[j] = __expf(s1[j] - mnew);
      float r = p0[j] + p1[j];
      r += __shfl_xor(r, 1, 32);
      r += __shfl_xor(r, 2, 32);
      r += __shfl_xor(r, 4, 32);
      r += __shfl_xor(r, 8, 32);
      li[j] = li[j] * sc + r;
      o0[j] *= sc;
      o1[j] *= sc;
      mi[j] = mnew;
    }

    // P: D-layout -> LDS row-major [16 x 32] bf16
#pragma unroll
    for (int j = 0; j < 8; ++j) {
      const int row = j + 8 * half;
      pl[row * 32 + ln]      = (__bf16)p0[j];
      pl[row * 32 + 16 + ln] = (__bf16)p1[j];
    }
    __syncthreads();

    // P A-fragment + V B-fragments (contraction dim = kv, 32 wide)
    v16bf pa, vb0, vb1;
#pragma unroll
    for (int i = 0; i < 16; ++i) {
      const int kk = akidx(half, i);
      pa[i]  = pl[ln * 32 + kk];
      vb0[i] = vw[(size_t)(b * Kk + kv0 + kk) * HC + h * Cc + ln];
      vb1[i] = vw[(size_t)(b * Kk + kv0 + kk) * HC + h * Cc + 16 + ln];
    }
    o0 = wmma_bf16(pa, vb0, o0);
    o1 = wmma_bf16(pa, vb1, o1);
    __syncthreads();
  }

  // epilogue: normalize, gate, store bf16 for the output projection
#pragma unroll
  for (int j = 0; j < 8; ++j) {
    const int row = q0 + j + 8 * half;
    const float inv = 1.0f / li[j];
    const size_t base = (size_t)(b * Qq + row) * HC + h * Cc;
    const float g0 = gate[base + ln];
    const float g1 = gate[base + 16 + ln];
    ow[base + ln]      = (__bf16)(o0[j] * inv * g0);
    ow[base + 16 + ln] = (__bf16)(o1[j] * inv * g1);
  }
}

// -----------------------------------------------------------------------------
extern "C" void kernel_launch(void* const* d_in, const int* in_sizes, int n_in,
                              void* d_out, int out_size, void* d_ws, size_t ws_size,
                              hipStream_t stream) {
  const float* q_x       = (const float*)d_in[0];
  const float* kv_x      = (const float*)d_in[1];
  const float* bias_mask = (const float*)d_in[2];
  const float* bias_pair = (const float*)d_in[3];
  const float* wq        = (const float*)d_in[4];
  const float* wk        = (const float*)d_in[5];
  const float* wv        = (const float*)d_in[6];
  const float* wg        = (const float*)d_in[7];
  const float* bg        = (const float*)d_in[8];
  const float* wo        = (const float*)d_in[9];
  const float* bo        = (const float*)d_in[10];
  float* out = (float*)d_out;

  // workspace layout (bytes)
  char* ws = (char*)d_ws;
  __bf16* q_ws = (__bf16*)(ws + 0);                  // 2 MB  [B*Q, HC]
  __bf16* k_ws = (__bf16*)(ws + (2ull << 20));       // 2 MB  [B*K, HC]
  __bf16* v_ws = (__bf16*)(ws + (4ull << 20));       // 2 MB  [B*K, HC]
  float*  g_ws = (float*) (ws + (6ull << 20));       // 4 MB  [B*Q, HC] f32
  __bf16* o_ws = (__bf16*)(ws + (10ull << 20));      // 2 MB  [B*Q, HC]

  const float qscale = 0.17677669529663689f; // 1/sqrt(C), C=32

  const dim3 gblk(128);
  const dim3 ggrd(Bb * Qq / 16, HC / 64); // 256 x 4

  // projections
  gemm16_wmma<0><<<ggrd, gblk, 0, stream>>>(q_x,  wq, nullptr, q_ws, Bb * Qq, HC, Dd, qscale);
  gemm16_wmma<1><<<ggrd, gblk, 0, stream>>>(kv_x, wk, nullptr, k_ws, Bb * Kk, HC, Dd, 1.0f);
  gemm16_wmma<1><<<ggrd, gblk, 0, stream>>>(kv_x, wv, nullptr, v_ws, Bb * Kk, HC, Dd, 1.0f);
  gemm16_wmma<3><<<ggrd, gblk, 0, stream>>>(q_x,  wg, bg,      g_ws, Bb * Qq, HC, Dd, 1.0f);

  // flash attention with biases + gating
  attn_flash_wmma<<<dim3(Qq / 16, Bb * Hh), dim3(32), 0, stream>>>(
      q_ws, k_ws, v_ws, g_ws, bias_mask, bias_pair, o_ws);

  // output projection
  gemm16_wmma<4><<<ggrd, gblk, 0, stream>>>(o_ws, wo, bo, out, Bb * Qq, Dd, HC, 1.0f);
}
